// HierarchicalBivectorBlock_84825604096574
// MI455X (gfx1250) — compile-verified, matrix operand order checked by reading, not run on hardware
//
#include <hip/hip_runtime.h>
#include <hip/hip_bf16.h>

typedef __attribute__((ext_vector_type(16))) _Float16 v16h;
typedef __attribute__((ext_vector_type(8)))  _Float16 v8h;
typedef __attribute__((ext_vector_type(8)))  float    v8f;
typedef __attribute__((ext_vector_type(4)))  unsigned int u32x4;
typedef __attribute__((ext_vector_type(8)))  int      i32x8;
typedef __attribute__((ext_vector_type(4)))  int      i32x4;

#define DEV __device__ __forceinline__

#define DIM   256
#define KPH   64
#define SEQL  1024
#define NTOK  (2 * SEQL)
#define PI_F  3.14159265358979f

#if defined(__has_builtin)
#if __has_builtin(__builtin_amdgcn_tensor_load_to_lds) && __has_builtin(__builtin_amdgcn_s_wait_tensorcnt)
#define HAS_TDM 1
#endif
#endif
#ifndef HAS_TDM
#define HAS_TDM 0
#endif

// ---------------------------------------------------------------------------
// Fragment loader (wave32, v_wmma_f32_16x16x32_f16 operand). Serves A and B:
//   A: row-major M x K  -> row = m = lane&15, K contiguous
//   B: stored   N x K   -> row = n = lane&15, K contiguous
// lane (row, h) reads halves [kbase+h*8,+8) and [kbase+16+h*8,+8): 2x ds_load_b128.
// ---------------------------------------------------------------------------
DEV v16h frag_ld(const _Float16* p, int stride, int kbase) {
  const int lane = threadIdx.x & 31;
  const _Float16* q = p + (lane & 15) * stride + kbase + (lane >> 4) * 8;
  v8h lo = *(const v8h*)(q);
  v8h hi = *(const v8h*)(q + 16);
  return __builtin_shufflevector(lo, hi, 0, 1, 2, 3, 4, 5, 6, 7,
                                 8, 9, 10, 11, 12, 13, 14, 15);
}

DEV v8f wmma_f(v16h a, v16h b, v8f c) {
  return __builtin_amdgcn_wmma_f32_16x16x32_f16(
      false, a, false, b, (short)0, c, false, false);
}

DEV float gelu_exact(float v) { return 0.5f * v * (1.0f + erff(v * 0.70710678118654752f)); }

// ---------------------------------------------------------------------------
// Contiguous f16 global -> LDS stage. TDM (tensor_load_to_lds) on wave 0 when
// available, else cooperative b128 copies. Caller brackets with __syncthreads.
// D# per CDNA5 ISA ch.8: group0 = {count=1, lds_addr, global_addr, type=2},
// group1 = {data_size=8B, 1-D tile of `halves/4` 8-byte units}.
// ---------------------------------------------------------------------------
DEV void stage_copy(_Float16* ldst, const _Float16* __restrict__ gsrc, int halves) {
#if HAS_TDM
  if ((threadIdx.x >> 5) == 0) {
    unsigned long long ga = (unsigned long long)(size_t)gsrc;
    unsigned int la = (unsigned int)(size_t)(void*)ldst;  // low 32b = LDS offset
    unsigned int n8 = (unsigned int)(halves >> 2);        // 8-byte units
    u32x4 g0;
    g0[0] = 1u;                                              // count=1 (user D#)
    g0[1] = la;                                              // lds_addr
    g0[2] = (unsigned int)ga;                                // global_addr[31:0]
    g0[3] = ((unsigned int)(ga >> 32) & 0x01FFFFFFu) | 0x80000000u;  // [56:32]|type=2
    i32x8 g1;
    g1[0] = (int)(3u << 16);                                 // data_size = 8 bytes
    g1[1] = (int)((n8 & 0xFFFFu) << 16);                     // tensor_dim0 lo
    g1[2] = (int)(((n8 >> 16) & 0xFFFFu) | (1u << 16));      // dim0 hi | tensor_dim1=1
    g1[3] = (int)((n8 & 0xFFFFu) << 16);                     // tile_dim0
    g1[4] = 1;                                               // tile_dim1 = 1
    g1[5] = (int)n8;                                         // tensor_dim0_stride
    g1[6] = 0;
    g1[7] = 0;
    i32x4 z4 = {0, 0, 0, 0};
    i32x8 z8 = {0, 0, 0, 0, 0, 0, 0, 0};
    __builtin_amdgcn_tensor_load_to_lds(g0, g1, z4, z4, z8, 0);
    __builtin_amdgcn_s_wait_tensorcnt(0);
  }
#else
  for (int idx = threadIdx.x * 8; idx < halves; idx += 2048)
    *(v8h*)(ldst + idx) = *(const v8h*)(gsrc + idx);
#endif
}

// ---------------- Kernel 0: one-time weight prep -----------------------------
// f32 W[K][N] -> f16 panel-transposed T[kp][n][32] (kr contiguous per n).
__global__ __launch_bounds__(256) void prep_kernel(
    const float* __restrict__ Wk1, const float* __restrict__ Wq1,
    const float* __restrict__ Wv,  const float* __restrict__ Wo,
    const float* __restrict__ Wk2, const float* __restrict__ Wq2,
    _Float16* Tk1, _Float16* Tq1, _Float16* Tv, _Float16* To,
    _Float16* Tk2, _Float16* Tq2) {
  int bidx = blockIdx.x;
  const float* W;
  _Float16* T;
  int ldw, kp;
  if (bidx < 32) {
    int w = bidx >> 3;
    kp = bidx & 7;
    ldw = DIM;
    W = (w == 0) ? Wk1 : (w == 1) ? Wq1 : (w == 2) ? Wv : Wo;
    T = (w == 0) ? Tk1 : (w == 1) ? Tq1 : (w == 2) ? Tv : To;
  } else {
    int w = (bidx - 32) >> 3;
    kp = (bidx - 32) & 7;
    ldw = KPH;
    W = (w == 0) ? Wk2 : Wq2;
    T = (w == 0) ? Tk2 : Tq2;
  }
  const int total = 32 * ldw;
  _Float16* dst = T + kp * total;
  const float* src = W + kp * 32 * ldw;
  for (int idx = threadIdx.x; idx < total; idx += 256) {
    int kr = idx / ldw, n = idx - kr * ldw;
    dst[n * 32 + kr] = (_Float16)src[kr * ldw + n];
  }
}

// ---------------- Kernel 1: encoders + V ------------------------------------
// Outputs (all f16): VT per-tile [c][16], KS/QF rows [token][cos|sin] (128).
__global__ __launch_bounds__(256) void encode_kernel(
    const float* __restrict__ x,
    const _Float16* __restrict__ Tk1, const float* __restrict__ bk1,
    const _Float16* __restrict__ Tk2, const float* __restrict__ bk2,
    const _Float16* __restrict__ Tq1, const float* __restrict__ bq1,
    const _Float16* __restrict__ Tq2, const float* __restrict__ bq2,
    const _Float16* __restrict__ Tv,  const float* __restrict__ bv,
    _Float16* __restrict__ VT, _Float16* __restrict__ KS,
    _Float16* __restrict__ QF) {
  __shared__ __align__(16) _Float16 lds_x[16 * DIM];
  __shared__ __align__(16) _Float16 lds_w[DIM * 32];
  __shared__ __align__(16) _Float16 lds_h[16 * DIM];

  const int tid  = threadIdx.x;
  const int wave = tid >> 5;
  const int lane = tid & 31;
  const int half = lane >> 4;
  const int nl   = lane & 15;
  const int tile = blockIdx.x;
  const int row0 = tile * 16;

  for (int idx = tid * 4; idx < 16 * DIM; idx += 1024) {
    float4 v = *(const float4*)(x + row0 * DIM + idx);
    lds_x[idx + 0] = (_Float16)v.x;
    lds_x[idx + 1] = (_Float16)v.y;
    lds_x[idx + 2] = (_Float16)v.z;
    lds_x[idx + 3] = (_Float16)v.w;
  }
  __syncthreads();

  // (16 x 256) = Asrc(16x256) @ W(256x256): 8 waves x two 16-col tiles.
  auto gemm256 = [&](const _Float16* Asrc, const _Float16* __restrict__ T,
                     v8f& acc0, v8f& acc1) {
    acc0 = {};
    acc1 = {};
    for (int kp = 0; kp < 8; ++kp) {
      __syncthreads();
      stage_copy(lds_w, T + kp * (DIM * 32), DIM * 32);
      __syncthreads();
      v16h a  = frag_ld(Asrc, DIM, kp * 32);
      v16h b0 = frag_ld(lds_w + (wave * 32) * 32, 32, 0);
      acc0 = wmma_f(a, b0, acc0);
      v16h b1 = frag_ld(lds_w + (wave * 32 + 16) * 32, 32, 0);
      acc1 = wmma_f(a, b1, acc1);
    }
  };

  // ---- V = x @ Wv + bv -> VT[tile][c][16] (f16, transposed tile) ----
  {
    v8f a0, a1;
    gemm256(lds_x, Tv, a0, a1);
    _Float16* vt = VT + tile * (DIM * 16);
#pragma unroll
    for (int r = 0; r < 8; ++r) {
      int m  = r + 8 * half;
      int c0 = wave * 32 + nl;
      vt[c0 * 16 + m]        = (_Float16)(a0[r] + bv[c0]);
      vt[(c0 + 16) * 16 + m] = (_Float16)(a1[r] + bv[c0 + 16]);
    }
  }

  // ---- encode(): tanh(gelu(x@W1+b1)@W2+b2) -> phase -> cos|sin rows ----
  auto encode = [&](const _Float16* __restrict__ T1, const float* __restrict__ b1,
                    const _Float16* __restrict__ T2, const float* __restrict__ b2,
                    _Float16* __restrict__ Po) {
    v8f a0, a1;
    gemm256(lds_x, T1, a0, a1);
    __syncthreads();  // protect lds_h from previous consumer
#pragma unroll
    for (int r = 0; r < 8; ++r) {
      int m  = r + 8 * half;
      int c0 = wave * 32 + nl;
      lds_h[m * DIM + c0]      = (_Float16)gelu_exact(a0[r] + b1[c0]);
      lds_h[m * DIM + c0 + 16] = (_Float16)gelu_exact(a1[r] + b1[c0 + 16]);
    }
    __syncthreads();
    v8f p = {};
    for (int kp = 0; kp < 8; ++kp) {
      __syncthreads();
      stage_copy(lds_w, T2 + kp * (KPH * 32), KPH * 32);
      __syncthreads();
      if (wave < 4) {
        v16h a = frag_ld(lds_h, DIM, kp * 32);
        v16h b = frag_ld(lds_w + (wave * 16) * 32, 32, 0);
        p = wmma_f(a, b, p);
      }
    }
    if (wave < 4) {
#pragma unroll
      for (int r = 0; r < 8; ++r) {
        int m = r + 8 * half;
        int c = wave * 16 + nl;
        float ph = tanhf(p[r] + b2[c]);
        float fr = exp2f((float)(c >> 4));  // 2^scale, 16 phases per scale
        ph *= fr * PI_F;
        float sv, cv;
        sincosf(ph, &sv, &cv);
        _Float16* d = Po + (row0 + m) * 128;
        d[c]      = (_Float16)cv;
        d[c + 64] = (_Float16)sv;
      }
    }
  };

  encode(Tk1, bk1, Tk2, bk2, KS);
  encode(Tq1, bq1, Tq2, bq2, QF);
}

// ---------------- Kernel 2: causal phasor attention --------------------------
// retrieved[l,d] = (1/sqrt((l+1)K)) * sum_{t<=l} (cq[l]·C[t] + sq[l]·S[t]) V[t,d]
__global__ __launch_bounds__(256) void attn_kernel(
    const _Float16* __restrict__ KS, const _Float16* __restrict__ QF,
    const _Float16* __restrict__ VT, float* __restrict__ R) {
  __shared__ __align__(16) _Float16 ldsQ[16 * 128];
  __shared__ __align__(16) _Float16 ldsK[16 * 128];
  __shared__ __align__(16) _Float16 ldsV[DIM * 32];  // N x 32, k 16..31 zero
  __shared__ __align__(16) _Float16 ldsS[16 * 32];   // M x 32, k 16..31 zero

  const int tid  = threadIdx.x;
  const int wave = tid >> 5;
  const int lane = tid & 31;
  const int half = lane >> 4;
  const int nl   = lane & 15;
  const int b    = blockIdx.x >> 6;
  const int qt   = blockIdx.x & 63;
  const int qrow0 = b * SEQL + qt * 16;

  // zero K-pad regions once
  for (int idx = tid; idx < DIM * 16; idx += 256)
    ldsV[(idx >> 4) * 32 + 16 + (idx & 15)] = (_Float16)0.0f;
  for (int idx = tid; idx < 16 * 16; idx += 256)
    ldsS[(idx >> 4) * 32 + 16 + (idx & 15)] = (_Float16)0.0f;

  stage_copy(ldsQ, QF + qrow0 * 128, 16 * 128);

  v8f acc0 = {}, acc1 = {};

  for (int kt = 0; kt <= qt; ++kt) {
    const int ktile = b * 64 + kt;
    __syncthreads();
    stage_copy(ldsK, KS + (ktile * 16) * 128, 16 * 128);
    {  // V tile: contiguous [c][16] rows -> LDS stride-32 rows (pure b128 copies)
      const _Float16* src = VT + ktile * (DIM * 16) + tid * 16;
      _Float16* dst = ldsV + tid * 32;
      *(v8h*)dst = *(const v8h*)src;
      *(v8h*)(dst + 8) = *(const v8h*)(src + 8);
    }
    __syncthreads();

    if (wave == 0) {  // 16x16 score tile = Q(16x128) x K^T(128x16)
      v8f sc = {};
#pragma unroll
      for (int ks = 0; ks < 4; ++ks) {
        v16h a  = frag_ld(ldsQ, 128, ks * 32);
        v16h bb = frag_ld(ldsK, 128, ks * 32);
        sc = wmma_f(a, bb, sc);
      }
#pragma unroll
      for (int r = 0; r < 8; ++r) {
        int m = r + 8 * half;
        float v = sc[r];
        if (kt == qt && nl > m) v = 0.0f;  // causal mask on diagonal tile
        ldsS[m * 32 + nl] = (_Float16)v;
      }
    }
    __syncthreads();

    v16h a  = frag_ld(ldsS, 32, 0);
    v16h b0 = frag_ld(ldsV + (wave * 32) * 32, 32, 0);
    acc0 = wmma_f(a, b0, acc0);
    v16h b1 = frag_ld(ldsV + (wave * 32 + 16) * 32, 32, 0);
    acc1 = wmma_f(a, b1, acc1);
  }

#pragma unroll
  for (int r = 0; r < 8; ++r) {
    int m = r + 8 * half;
    float scale = rsqrtf((float)(qt * 16 + m + 1) * (float)KPH);
    int c = wave * 32 + nl;
    R[(qrow0 + m) * DIM + c]      = acc0[r] * scale;
    R[(qrow0 + m) * DIM + c + 16] = acc1[r] * scale;
  }
}

// ---------------- Kernel 3: layernorm + out projection + residual -----------
__global__ __launch_bounds__(256) void out_kernel(
    const float* __restrict__ x,    const float* __restrict__ R,
    const float* __restrict__ ln_w, const float* __restrict__ ln_b,
    const _Float16* __restrict__ To, const float* __restrict__ bo,
    float* __restrict__ out) {
  __shared__ __align__(16) float    lds_r[16 * DIM];
  __shared__ __align__(16) _Float16 lds_n[16 * DIM];
  __shared__ __align__(16) _Float16 lds_w[DIM * 32];

  const int tid  = threadIdx.x;
  const int wave = tid >> 5;
  const int lane = tid & 31;
  const int half = lane >> 4;
  const int nl   = lane & 15;
  const int row0 = blockIdx.x * 16;

  for (int idx = tid * 4; idx < 16 * DIM; idx += 1024)
    *(float4*)(lds_r + idx) = *(const float4*)(R + row0 * DIM + idx);
  __syncthreads();

  for (int rr = 0; rr < 2; ++rr) {
    int row = wave * 2 + rr;
    float sum = 0.0f, sq = 0.0f;
    for (int j = lane; j < DIM; j += 32) {
      float v = lds_r[row * DIM + j];
      sum += v;
      sq += v * v;
    }
    for (int off = 16; off > 0; off >>= 1) {
      sum += __shfl_xor(sum, off, 32);
      sq  += __shfl_xor(sq, off, 32);
    }
    float mu   = sum * (1.0f / DIM);
    float var  = sq * (1.0f / DIM) - mu * mu;
    float rstd = rsqrtf(var + 1e-5f);
    for (int j = lane; j < DIM; j += 32) {
      float v = (lds_r[row * DIM + j] - mu) * rstd * ln_w[j] + ln_b[j];
      lds_n[row * DIM + j] = (_Float16)v;
    }
  }
  __syncthreads();

  v8f a0 = {}, a1 = {};
  for (int kp = 0; kp < 8; ++kp) {
    __syncthreads();
    stage_copy(lds_w, To + kp * (DIM * 32), DIM * 32);
    __syncthreads();
    v16h a  = frag_ld(lds_n, DIM, kp * 32);
    v16h b0 = frag_ld(lds_w + (wave * 32) * 32, 32, 0);
    a0 = wmma_f(a, b0, a0);
    v16h b1 = frag_ld(lds_w + (wave * 32 + 16) * 32, 32, 0);
    a1 = wmma_f(a, b1, a1);
  }
#pragma unroll
  for (int r = 0; r < 8; ++r) {
    int m = r + 8 * half;
    int c = wave * 32 + nl;
    out[(row0 + m) * DIM + c]      = x[(row0 + m) * DIM + c]      + a0[r] + bo[c];
    out[(row0 + m) * DIM + c + 16] = x[(row0 + m) * DIM + c + 16] + a1[r] + bo[c + 16];
  }
}

// ---------------- Launch -----------------------------------------------------
extern "C" void kernel_launch(void* const* d_in, const int* in_sizes, int n_in,
                              void* d_out, int out_size, void* d_ws, size_t ws_size,
                              hipStream_t stream) {
  const float* x    = (const float*)d_in[0];
  const float* Wk1  = (const float*)d_in[1];
  const float* bk1  = (const float*)d_in[2];
  const float* Wk2  = (const float*)d_in[3];
  const float* bk2  = (const float*)d_in[4];
  const float* Wq1  = (const float*)d_in[5];
  const float* bq1  = (const float*)d_in[6];
  const float* Wq2  = (const float*)d_in[7];
  const float* bq2  = (const float*)d_in[8];
  const float* Wv   = (const float*)d_in[9];
  const float* bv   = (const float*)d_in[10];
  const float* ln_w = (const float*)d_in[11];
  const float* ln_b = (const float*)d_in[12];
  const float* Wo   = (const float*)d_in[13];
  const float* bo   = (const float*)d_in[14];

  // f16 workspace layout (all 16B aligned)
  _Float16* Tk1 = (_Float16*)d_ws;         // 8 panels x 256 x 32
  _Float16* Tq1 = Tk1 + DIM * DIM;
  _Float16* Tv  = Tq1 + DIM * DIM;
  _Float16* To  = Tv + DIM * DIM;
  _Float16* Tk2 = To + DIM * DIM;          // 8 panels x 64 x 32
  _Float16* Tq2 = Tk2 + DIM * KPH;
  _Float16* KS  = Tq2 + DIM * KPH;         // NTOK x 128  (cos|sin keys)
  _Float16* QF  = KS + NTOK * 128;         // NTOK x 128  (cos|sin queries)
  _Float16* VT  = QF + NTOK * 128;         // 128 tiles x 256 x 16
  float*    R   = (float*)(VT + NTOK * DIM);  // NTOK x 256 f32

  const int tiles = NTOK / 16;  // 128
  prep_kernel<<<48, 256, 0, stream>>>(Wk1, Wq1, Wv, Wo, Wk2, Wq2,
                                      Tk1, Tq1, Tv, To, Tk2, Tq2);
  encode_kernel<<<tiles, 256, 0, stream>>>(x, Tk1, bk1, Tk2, bk2, Tq1, bq1,
                                           Tq2, bq2, Tv, bv, VT, KS, QF);
  attn_kernel<<<tiles, 256, 0, stream>>>(KS, QF, VT, R);
  out_kernel<<<tiles, 256, 0, stream>>>(x, R, ln_w, ln_b, To, bo, (float*)d_out);
}
